// LiquidTimeConstantNetwork_6725918785575
// MI455X (gfx1250) — matrix operand order repaired and checked by reading, not compiled, not run on hardware
//
#include <hip/hip_runtime.h>

typedef __attribute__((ext_vector_type(16))) __bf16 v16bf;
typedef __attribute__((ext_vector_type(8)))  float  v8f;

#define B_ 64
#define S_ 1024
#define I_ 128
#define H_ 256
#define O_ 128
#define DT_ 0.1f

// ---- dynamic LDS layout (bytes) ----
#define HBF_OFF   0            // h state bf16            [16][256]  ->  8192 B
#define CG_OFF    8192         // cg preact f32           [16][256]  -> 16384 B
#define EG_OFF    24576        // eg preact f32           [16][256]  -> 16384 B
#define WEG_OFF   40960        // W_eg frags: 16 nt * 8 kk * 1KB     -> 131072 B
#define WIN_OFF   172032       // W_in frags: 16 nt * 4 kk * 1KB     ->  65536 B
#define WOUT_OFF  237568       // W_out frags: 8 nt * 8 kk * 1KB     ->  65536 B
#define SMEM_SZ   303104

union FragU { unsigned int u[8]; v16bf v; };

__device__ __forceinline__ unsigned int pack2_bf16(float lo, float hi) {
  unsigned int a = __float_as_uint(lo);
  unsigned int b = __float_as_uint(hi);
  a = (a + 0x7FFFu + ((a >> 16) & 1u)) >> 16;   // RNE
  b = (b + 0x7FFFu + ((b >> 16) & 1u)) >> 16;
  return (a & 0xFFFFu) | (b << 16);
}

__device__ __forceinline__ unsigned short f2bf(float f) {
  unsigned int a = __float_as_uint(f);
  return (unsigned short)((a + 0x7FFFu + ((a >> 16) & 1u)) >> 16);
}

// Build a B fragment (32x16 bf16, KxN) from row-major f32 W[Ntot][Ktot]; B[k][n] = W[n][k].
__device__ __forceinline__ FragU make_b_frag(const float* W, int ldk, int ntile, int kk, int lane) {
  FragU f;
  const int n  = ntile * 16 + (lane & 15);
  const int kb = (lane >> 4) << 3;               // lanes 16-31 -> K += 8
#pragma unroll
  for (int j = 0; j < 8; ++j) {
    const int k = kk * 32 + kb + 2 * j + ((j >= 4) ? 8 : 0);
    const float2 p = *(const float2*)(W + (size_t)n * ldk + k);
    f.u[j] = pack2_bf16(p.x, p.y);
  }
  return f;
}

// Swizzled fragment store/load in LDS: each lane's 8 dwords contiguous (2x b128).
__device__ __forceinline__ void store_frag_lds(char* base, int fragIdx, int lane, const FragU& f) {
  uint4* p = (uint4*)(base + fragIdx * 1024 + lane * 32);
  p[0] = make_uint4(f.u[0], f.u[1], f.u[2], f.u[3]);
  p[1] = make_uint4(f.u[4], f.u[5], f.u[6], f.u[7]);
}

__device__ __forceinline__ v16bf load_frag_lds(const char* base, int fragIdx, int lane) {
  FragU f;
  const uint4* p = (const uint4*)(base + fragIdx * 1024 + lane * 32);
  const uint4 a = p[0], b = p[1];
  f.u[0] = a.x; f.u[1] = a.y; f.u[2] = a.z; f.u[3] = a.w;
  f.u[4] = b.x; f.u[5] = b.y; f.u[6] = b.z; f.u[7] = b.w;
  return f.v;
}

// A fragment (16x32 bf16, MxK) from LDS-resident bf16 h[16][H_].
__device__ __forceinline__ v16bf load_a_h(const unsigned short* hbf, int kk, int lane) {
  FragU f;
  const int m  = lane & 15;
  const int kb = (lane >> 4) << 3;               // lanes 16-31 -> K += 8
#pragma unroll
  for (int j = 0; j < 8; ++j) {
    const int k = kk * 32 + kb + 2 * j + ((j >= 4) ? 8 : 0);
    f.u[j] = *(const unsigned int*)(hbf + m * H_ + k);
  }
  return f.v;
}

// A fragment from global f32 x rows (row stride S_*I_), cvt->bf16 inline.
__device__ __forceinline__ v16bf load_a_x(const float* xbase, int kk, int lane) {
  FragU f;
  const int m  = lane & 15;
  const int kb = (lane >> 4) << 3;
  const float* row = xbase + (size_t)m * (size_t)(S_ * I_);
#pragma unroll
  for (int j = 0; j < 8; ++j) {
    const int k = kk * 32 + kb + 2 * j + ((j >= 4) ? 8 : 0);
    const float2 p = *(const float2*)(row + k);
    f.u[j] = pack2_bf16(p.x, p.y);
  }
  return f.v;
}

__global__ __launch_bounds__(1024, 1) void ltcn_persistent(
    const float* __restrict__ x,
    const float* __restrict__ W_in,  const float* __restrict__ b_in,
    const float* __restrict__ W_rec, const float* __restrict__ b_rec,
    const float* __restrict__ W_out, const float* __restrict__ b_out,
    const float* __restrict__ W_cg,  const float* __restrict__ b_cg,
    const float* __restrict__ W_eg,  const float* __restrict__ b_eg,
    const float* __restrict__ tau,
    float* __restrict__ out, float* __restrict__ cs)
{
  extern __shared__ char smem[];
  unsigned short* hbf = (unsigned short*)(smem + HBF_OFF);
  float* cg_s = (float*)(smem + CG_OFF);
  float* eg_s = (float*)(smem + EG_OFF);

  const int tid  = threadIdx.x;
  const int wave = tid >> 5;
  const int lane = tid & 31;
  const int b0   = blockIdx.x * 16;
  const int lmod = lane & 15;
  const int lhi  = lane >> 4;

  // ---- one-time: zero h0, swizzle W_eg / W_in / W_out into LDS fragment layout ----
  for (int i = tid; i < 16 * H_; i += 1024) hbf[i] = 0;
  for (int f = wave; f < 256; f += 32) {
    if (f < 128) {                     // W_eg: 16 nt x 8 kk
      const int nt = f >> 3, kk = f & 7;
      store_frag_lds(smem + WEG_OFF, f, lane, make_b_frag(W_eg, H_, nt, kk, lane));
    } else if (f < 192) {              // W_in: 16 nt x 4 kk
      const int g = f - 128, nt = g >> 2, kk = g & 3;
      store_frag_lds(smem + WIN_OFF, g, lane, make_b_frag(W_in, I_, nt, kk, lane));
    } else {                           // W_out: 8 nt x 8 kk
      const int g = f - 192, nt = g >> 3, kk = g & 7;
      store_frag_lds(smem + WOUT_OFF, g, lane, make_b_frag(W_out, H_, nt, kk, lane));
    }
  }
  __syncthreads();

  if (wave < 16) {
    // ================= R-waves: inp + rec GEMM, liquid state update =================
    const int nt = wave;
    const int n  = nt * 16 + lmod;
    const float bias_ir = b_in[n] + b_rec[n];
    const float itau    = 1.0f / tau[n];

    v16bf wr[8];                                   // W_rec tile: register-resident
#pragma unroll
    for (int kk = 0; kk < 8; ++kk) wr[kk] = make_b_frag(W_rec, H_, nt, kk, lane).v;

    float hreg[8];
#pragma unroll
    for (int r = 0; r < 8; ++r) hreg[r] = 0.0f;

    for (int s = 0; s < S_; ++s) {
      const float* xbase = x + ((size_t)b0 * S_ + (size_t)s) * I_;
      if (s + 1 < S_)
        __builtin_prefetch(x + ((size_t)(b0 + lmod) * S_ + (size_t)(s + 1)) * I_, 0, 0);

      v8f acc;
#pragma unroll
      for (int r = 0; r < 8; ++r) acc[r] = bias_ir;
#pragma unroll
      for (int kk = 0; kk < 4; ++kk) {             // x_t @ W_in.T (W_in frags from LDS)
        v16bf a = load_a_x(xbase, kk, lane);
        v16bf b = load_frag_lds(smem + WIN_OFF, nt * 4 + kk, lane);
        acc = __builtin_amdgcn_wmma_f32_16x16x32_bf16(false, a, false, b,
                                                      (short)0, acc, false, false);
      }
#pragma unroll
      for (int kk = 0; kk < 8; ++kk) {             // h @ W_rec.T (register B)
        v16bf a = load_a_h(hbf, kk, lane);
        acc = __builtin_amdgcn_wmma_f32_16x16x32_bf16(false, a, false, wr[kk],
                                                      (short)0, acc, false, false);
      }
      __syncthreads();                             // (1) gates published; h reads done

#pragma unroll
      for (int r = 0; r < 8; ++r) {                // fp32 liquid update
        const int m   = r + (lhi << 3);
        const float cg  = cg_s[m * H_ + n];
        const float eg  = eg_s[m * H_ + n];
        const float pre = fmaxf(acc[r], 0.0f);
        const float cm  = 1.0f / (1.0f + __expf(-cg));
        float dh = (pre - hreg[r]) * itau;
        dh = dh * cm + 0.1f * tanhf(eg);
        const float hn = hreg[r] + DT_ * dh;
        hreg[r] = hn;
        hbf[m * H_ + n] = f2bf(hn);
        cs[((size_t)(b0 + m) * S_ + (size_t)s) * H_ + n] = hn;
      }
      __syncthreads();                             // (2) new h visible
    }
  } else {
    // ================= G-waves: cg (register B) + eg (LDS B); waves 24-31 out-proj ===
    const int nt = wave - 16;
    const int n  = nt * 16 + lmod;
    const float bcg = b_cg[n];
    const float beg = b_eg[n];

    v16bf wc[8];                                   // W_cg tile: register-resident
#pragma unroll
    for (int kk = 0; kk < 8; ++kk) wc[kk] = make_b_frag(W_cg, H_, nt, kk, lane).v;

    const int  ot     = wave - 24;                 // 0..7 valid for waves 24-31
    const bool doOut  = (wave >= 24);
    const int  no     = (doOut ? ot : 0) * 16 + lmod;
    const float bias_o = doOut ? b_out[no] : 0.0f;

    for (int s = 0; s < S_; ++s) {
      v8f ac, ae;
#pragma unroll
      for (int r = 0; r < 8; ++r) { ac[r] = bcg; ae[r] = beg; }
#pragma unroll
      for (int kk = 0; kk < 8; ++kk) {
        v16bf a  = load_a_h(hbf, kk, lane);        // shared A for both gates
        v16bf be = load_frag_lds(smem + WEG_OFF, nt * 8 + kk, lane);
        ac = __builtin_amdgcn_wmma_f32_16x16x32_bf16(false, a, false, wc[kk],
                                                     (short)0, ac, false, false);
        ae = __builtin_amdgcn_wmma_f32_16x16x32_bf16(false, a, false, be,
                                                     (short)0, ae, false, false);
      }
#pragma unroll
      for (int r = 0; r < 8; ++r) {
        const int m = r + (lhi << 3);
        cg_s[m * H_ + n] = ac[r];
        eg_s[m * H_ + n] = ae[r];
      }
      __syncthreads();                             // (1)
      __syncthreads();                             // (2) new h visible

      if (doOut) {                                 // out = h_new @ W_out.T + b_out
        v8f oa;
#pragma unroll
        for (int r = 0; r < 8; ++r) oa[r] = bias_o;
#pragma unroll
        for (int kk = 0; kk < 8; ++kk) {
          v16bf a = load_a_h(hbf, kk, lane);
          v16bf b = load_frag_lds(smem + WOUT_OFF, ot * 8 + kk, lane);
          oa = __builtin_amdgcn_wmma_f32_16x16x32_bf16(false, a, false, b,
                                                       (short)0, oa, false, false);
        }
#pragma unroll
        for (int r = 0; r < 8; ++r) {
          const int m = r + (lhi << 3);
          out[((size_t)(b0 + m) * S_ + (size_t)s) * O_ + no] = oa[r];
        }
      }
    }
  }
}

extern "C" void kernel_launch(void* const* d_in, const int* in_sizes, int n_in,
                              void* d_out, int out_size, void* d_ws, size_t ws_size,
                              hipStream_t stream) {
  (void)in_sizes; (void)n_in; (void)out_size; (void)d_ws; (void)ws_size;
  const float* x     = (const float*)d_in[0];
  const float* W_in  = (const float*)d_in[1];
  const float* b_in  = (const float*)d_in[2];
  const float* W_rec = (const float*)d_in[3];
  const float* b_rec = (const float*)d_in[4];
  const float* W_out = (const float*)d_in[5];
  const float* b_out = (const float*)d_in[6];
  const float* W_cg  = (const float*)d_in[7];
  const float* b_cg  = (const float*)d_in[8];
  const float* W_eg  = (const float*)d_in[9];
  const float* b_eg  = (const float*)d_in[10];
  const float* tau   = (const float*)d_in[11];

  float* out = (float*)d_out;                      // [B,S,O]
  float* cs  = out + (size_t)B_ * S_ * O_;         // [B,S,H]

  static bool attrSet = false;
  if (!attrSet) {   // host-side attribute, not a stream op (graph-capture safe)
    hipFuncSetAttribute((const void*)ltcn_persistent,
                        hipFuncAttributeMaxDynamicSharedMemorySize, SMEM_SZ);
    attrSet = true;
  }

  ltcn_persistent<<<dim3(B_ / 16), dim3(1024), SMEM_SZ, stream>>>(
      x, W_in, b_in, W_rec, b_rec, W_out, b_out, W_cg, b_cg, W_eg, b_eg, tau, out, cs);
}
